// TextHierarchyGraphEncoder_1889785610907
// MI455X (gfx1250) — matrix operand order repaired and checked by reading, not compiled, not run on hardware
//
#include <hip/hip_runtime.h>
#include <math.h>

typedef __attribute__((ext_vector_type(16))) _Float16 v16h;
typedef __attribute__((ext_vector_type(8)))  _Float16 v8h;
typedef __attribute__((ext_vector_type(8)))  float    v8f;

#define NN   50000
#define DIM  512
#define NE   100000
#define NREL 4

// ---------------------------------------------------------------------------
// helpers
// ---------------------------------------------------------------------------
__device__ __forceinline__ float blockSum256(float v, float* sm) {
  #pragma unroll
  for (int off = 16; off > 0; off >>= 1) v += __shfl_xor(v, off, 32);
  int w = threadIdx.x >> 5, ln = threadIdx.x & 31;
  if (ln == 0) sm[w] = v;
  __syncthreads();
  if (threadIdx.x < 32) {
    float t = (ln < 8) ? sm[ln] : 0.f;
    #pragma unroll
    for (int off = 4; off > 0; off >>= 1) t += __shfl_xor(t, off, 32);
    if (ln == 0) sm[8] = t;
  }
  __syncthreads();
  float r = sm[8];
  __syncthreads();
  return r;
}

__device__ __forceinline__ v16h ld_frag16(const _Float16* p0, const _Float16* p1) {
  v8h a = *(const v8h*)p0;
  v8h b = *(const v8h*)p1;
  v16h r;
  #pragma unroll
  for (int i = 0; i < 8; i++) { r[i] = a[i]; r[i + 8] = b[i]; }
  return r;
}

// ---------------------------------------------------------------------------
// utility kernels
// ---------------------------------------------------------------------------
__global__ void zero_f32(float* __restrict__ p, long n) {
  long i = (long)blockIdx.x * blockDim.x + threadIdx.x;
  long s = (long)gridDim.x * blockDim.x;
  for (; i < n; i += s) p[i] = 0.f;
}

__global__ void cvt_f32_f16(const float* __restrict__ s, _Float16* __restrict__ d, long n) {
  long i = (long)blockIdx.x * blockDim.x + threadIdx.x;
  long st = (long)gridDim.x * blockDim.x;
  for (; i < n; i += st) d[i] = (_Float16)s[i];
}

// ---------------------------------------------------------------------------
// input layernorm + node-type embedding:  base = LN(nf + emb[type]) ; xh = fp16(base)
// ---------------------------------------------------------------------------
__global__ __launch_bounds__(256) void prep_ln(
    const float* __restrict__ nf, const int* __restrict__ ntype,
    const float* __restrict__ emb, const float* __restrict__ g, const float* __restrict__ b,
    float* __restrict__ base, _Float16* __restrict__ xh) {
  __shared__ float sm[9];
  int row = blockIdx.x, t = threadIdx.x;
  int d0 = t, d1 = t + 256;
  size_t ro = (size_t)row * DIM;
  int nt = ntype[row];
  float x0 = nf[ro + d0] + emb[(size_t)nt * DIM + d0];
  float x1 = nf[ro + d1] + emb[(size_t)nt * DIM + d1];
  float s  = blockSum256(x0 + x1, sm);
  float ss = blockSum256(x0 * x0 + x1 * x1, sm);
  float mu = s * (1.f / DIM);
  float var = ss * (1.f / DIM) - mu * mu;
  float inv = rsqrtf(var + 1e-5f);
  float y0 = (x0 - mu) * inv * g[d0] + b[d0];
  float y1 = (x1 - mu) * inv * g[d1] + b[d1];
  base[ro + d0] = y0; base[ro + d1] = y1;
  xh[ro + d0] = (_Float16)y0; xh[ro + d1] = (_Float16)y1;
}

// ---------------------------------------------------------------------------
// WMMA GEMM:  C[z] = A[M,K] @ B[z][N,K]^T + bias[z]     (A shared across z)
// tile 128x64, BK=64, double-buffered LDS, 8 waves, each wave -> 2x2 of 16x16
// v_wmma_f32_16x16x32_f16 (x2 K-halves = 8 WMMA per barrier).
// EPI: 0 = fp32 store, 1 = fp16 store, 2 = exact GELU then fp16 store
// ---------------------------------------------------------------------------
template <int EPI>
__global__ __launch_bounds__(256) void gemm_wmma(
    const _Float16* __restrict__ A, const _Float16* __restrict__ Ball,
    const float* __restrict__ biasAll, void* __restrict__ Cout,
    int M, int N, int K, long sB, long sBias, long sC) {
  constexpr int BM = 128, BN = 64, BK = 64, LDT = 72;
  __shared__ __align__(16) _Float16 As[2][BM * LDT];
  __shared__ __align__(16) _Float16 Bs[2][BN * LDT];

  const int z = blockIdx.z;
  const _Float16* Bz   = Ball    + (size_t)z * sB;
  const float*    bias = biasAll + (size_t)z * sBias;

  const int tid = threadIdx.x;
  const int lane = tid & 31, wid = tid >> 5;
  const int lr = lane & 15, lh = lane >> 4;     // lane row / half-select
  const int wm = (wid & 3) * 32;                // wave tile offsets
  const int wn = (wid >> 2) * 32;
  const int mBase = blockIdx.y * BM;
  const int nBase = blockIdx.x * BN;

  // staging chunk coordinates (8-half = 16B chunks)
  // A: 1024 chunks (BM*BK/8), 4 per thread ; B: 512 chunks, 2 per thread
  int arow[4], akh[4], brow[2], bkh[2];
  #pragma unroll
  for (int t = 0; t < 4; t++) {
    int c = tid + t * 256;
    arow[t] = c >> 3;               // 8 chunks per 64-half row
    akh[t]  = (c & 7) * 8;
  }
  #pragma unroll
  for (int t = 0; t < 2; t++) {
    int c = tid + t * 256;
    brow[t] = c >> 3;
    bkh[t]  = (c & 7) * 8;
  }

  v8f acc[2][2];
  #pragma unroll
  for (int i = 0; i < 2; i++)
    #pragma unroll
    for (int j = 0; j < 2; j++)
      #pragma unroll
      for (int r = 0; r < 8; r++) acc[i][j][r] = 0.f;

  v8h ra[4], rb[2];
  auto gload = [&](int k0) {
    #pragma unroll
    for (int t = 0; t < 4; t++) {
      int gm = mBase + arow[t]; if (gm >= M) gm = M - 1;  // clamp; stores guarded
      ra[t] = *(const v8h*)&A[(size_t)gm * K + k0 + akh[t]];
    }
    #pragma unroll
    for (int t = 0; t < 2; t++)
      rb[t] = *(const v8h*)&Bz[(size_t)(nBase + brow[t]) * K + k0 + bkh[t]];
  };
  auto lstore = [&](int buf) {
    #pragma unroll
    for (int t = 0; t < 4; t++)
      *(v8h*)&As[buf][arow[t] * LDT + akh[t]] = ra[t];
    #pragma unroll
    for (int t = 0; t < 2; t++)
      *(v8h*)&Bs[buf][brow[t] * LDT + bkh[t]] = rb[t];
  };

  const int nk = K / BK;
  gload(0);
  lstore(0);
  int cur = 0;

  for (int kt = 0; kt < nk; ++kt) {
    __syncthreads();                         // buf[cur] ready; buf[cur^1] drained
    if (kt + 1 < nk) gload((kt + 1) * BK);   // overlap with WMMAs below

    #pragma unroll
    for (int kk = 0; kk < 2; kk++) {         // two 32-wide K slices
      v16h af[2], bfr[2];
      #pragma unroll
      for (int i = 0; i < 2; i++) {
        const _Float16* pr = &As[cur][(wm + i * 16 + lr) * LDT + kk * 32];
        af[i] = ld_frag16(pr + lh * 8, pr + 16 + lh * 8);
      }
      #pragma unroll
      for (int j = 0; j < 2; j++) {
        const _Float16* pc = &Bs[cur][(wn + j * 16 + lr) * LDT + kk * 32];
        bfr[j] = ld_frag16(pc + lh * 8, pc + 16 + lh * 8);
      }
      #pragma unroll
      for (int i = 0; i < 2; i++)
        #pragma unroll
        for (int j = 0; j < 2; j++)
          acc[i][j] = __builtin_amdgcn_wmma_f32_16x16x32_f16(
              false, af[i], false, bfr[j], (short)0, acc[i][j], false, false);
    }

    if (kt + 1 < nk) lstore(cur ^ 1);
    cur ^= 1;
  }

  // epilogue: C/D layout => element r -> M = r + 8*lh, N = lr
  const bool fullTile = (mBase + BM <= M);
  #pragma unroll
  for (int i = 0; i < 2; i++) {
    #pragma unroll
    for (int j = 0; j < 2; j++) {
      int gcol = nBase + wn + j * 16 + lr;
      float bv = bias[gcol];
      int growb = mBase + wm + i * 16 + lh * 8;
      size_t base_idx = (size_t)z * sC + (size_t)growb * N + gcol;
      if (fullTile) {
        if (EPI == 0) {
          float* p = (float*)Cout + base_idx;
          #pragma unroll
          for (int r = 0; r < 8; r++) p[(size_t)r * N] = acc[i][j][r] + bv;
        } else {
          _Float16* p = (_Float16*)Cout + base_idx;
          #pragma unroll
          for (int r = 0; r < 8; r++) {
            float v = acc[i][j][r] + bv;
            if (EPI == 2) v = 0.5f * v * (1.f + erff(v * 0.70710678118654752f));
            p[(size_t)r * N] = (_Float16)v;
          }
        }
      } else {
        #pragma unroll
        for (int r = 0; r < 8; r++) {
          if (growb + r < M) {
            float v = acc[i][j][r] + bv;
            if (EPI == 2) v = 0.5f * v * (1.f + erff(v * 0.70710678118654752f));
            if (EPI == 0) ((float*)Cout)[base_idx + (size_t)r * N] = v;
            else          ((_Float16*)Cout)[base_idx + (size_t)r * N] = (_Float16)v;
          }
        }
      }
    }
  }
}

// ---------------------------------------------------------------------------
// edge scatter: agg[dst] += w * H[rel][src];  deg[dst] += w   (reverse edges too)
// ---------------------------------------------------------------------------
__global__ __launch_bounds__(256) void scatter_edges(
    const _Float16* __restrict__ H, const int* __restrict__ eidx,
    const int* __restrict__ etype, const float* __restrict__ ew,
    float* __restrict__ agg, float* __restrict__ deg) {
  int e = blockIdx.x;
  int rev = (e >= NE) ? 1 : 0;
  int eb = rev ? e - NE : e;
  int s = eidx[rev ? NE + eb : eb];
  int d = eidx[rev ? eb : NE + eb];
  int rel = etype[eb] + (rev ? 2 : 0);
  float w = ew[eb];
  const _Float16* hr = H + ((size_t)rel * NN + (size_t)s) * DIM;
  float* ar = agg + (size_t)d * DIM;
  for (int c = threadIdx.x; c < DIM; c += 256)
    unsafeAtomicAdd(&ar[c], w * (float)hr[c]);
  if (threadIdx.x == 0) unsafeAtomicAdd(&deg[d], w);
}

// ---------------------------------------------------------------------------
// combine: x1 = LN(base + selfF + agg/max(deg,1)) ; x1h fp16 copy
// ---------------------------------------------------------------------------
__global__ __launch_bounds__(256) void combine_ln(
    const float* __restrict__ base, const float* __restrict__ selfF,
    const float* __restrict__ agg, const float* __restrict__ deg,
    const float* __restrict__ g, const float* __restrict__ b,
    float* __restrict__ x1, _Float16* __restrict__ x1h) {
  __shared__ float sm[9];
  int row = blockIdx.x, t = threadIdx.x;
  int d0 = t, d1 = t + 256;
  size_t ro = (size_t)row * DIM;
  float dg = deg[row]; dg = (dg > 1.f) ? dg : 1.f;
  float inv_dg = 1.f / dg;
  float a0 = base[ro + d0] + selfF[ro + d0] + agg[ro + d0] * inv_dg;
  float a1 = base[ro + d1] + selfF[ro + d1] + agg[ro + d1] * inv_dg;
  float s  = blockSum256(a0 + a1, sm);
  float ss = blockSum256(a0 * a0 + a1 * a1, sm);
  float mu = s * (1.f / DIM);
  float var = ss * (1.f / DIM) - mu * mu;
  float inv = rsqrtf(var + 1e-5f);
  float y0 = (a0 - mu) * inv * g[d0] + b[d0];
  float y1 = (a1 - mu) * inv * g[d1] + b[d1];
  x1[ro + d0] = y0; x1[ro + d1] = y1;
  x1h[ro + d0] = (_Float16)y0; x1h[ro + d1] = (_Float16)y1;
}

// ---------------------------------------------------------------------------
// final: x2 = LN(x1 + ffn, n2) ; xo = LN(x2, out) ; out = base + gm*(xo - base)
// ---------------------------------------------------------------------------
__global__ __launch_bounds__(256) void final_ln(
    const float* __restrict__ x1, const float* __restrict__ f,
    const float* __restrict__ base,
    const float* __restrict__ n2g, const float* __restrict__ n2b,
    const float* __restrict__ og, const float* __restrict__ ob,
    const float* __restrict__ gml, float* __restrict__ out) {
  __shared__ float sm[9];
  int row = blockIdx.x, t = threadIdx.x;
  int d0 = t, d1 = t + 256;
  size_t ro = (size_t)row * DIM;
  float a0 = x1[ro + d0] + f[ro + d0];
  float a1 = x1[ro + d1] + f[ro + d1];
  float s  = blockSum256(a0 + a1, sm);
  float ss = blockSum256(a0 * a0 + a1 * a1, sm);
  float mu = s * (1.f / DIM), var = ss * (1.f / DIM) - mu * mu;
  float inv = rsqrtf(var + 1e-5f);
  float y0 = (a0 - mu) * inv * n2g[d0] + n2b[d0];
  float y1 = (a1 - mu) * inv * n2g[d1] + n2b[d1];
  s  = blockSum256(y0 + y1, sm);
  ss = blockSum256(y0 * y0 + y1 * y1, sm);
  mu = s * (1.f / DIM); var = ss * (1.f / DIM) - mu * mu;
  inv = rsqrtf(var + 1e-5f);
  float z0 = (y0 - mu) * inv * og[d0] + ob[d0];
  float z1 = (y1 - mu) * inv * og[d1] + ob[d1];
  float gm = 1.f / (1.f + __expf(-gml[0]));
  float b0 = base[ro + d0], b1 = base[ro + d1];
  out[ro + d0] = b0 + gm * (z0 - b0);
  out[ro + d1] = b1 + gm * (z1 - b1);
}

// ---------------------------------------------------------------------------
// column sums over output matrix (for mean pooling)
// ---------------------------------------------------------------------------
__global__ __launch_bounds__(256) void colsum_k(const float* __restrict__ x, float* __restrict__ cs) {
  int d0 = threadIdx.x, d1 = threadIdx.x + 256;
  float s0 = 0.f, s1 = 0.f;
  for (int r = blockIdx.x; r < NN; r += gridDim.x) {
    const float* xr = x + (size_t)r * DIM;
    s0 += xr[d0]; s1 += xr[d1];
  }
  unsafeAtomicAdd(&cs[d0], s0);
  unsafeAtomicAdd(&cs[d1], s1);
}

__global__ void pooled_k(const float* __restrict__ x, const float* __restrict__ cs,
                         const float* __restrict__ dml, float* __restrict__ outp) {
  int d = threadIdx.x;
  float dm = 1.f / (1.f + __expf(-dml[0]));
  outp[d] = dm * x[d] + (1.f - dm) * cs[d] * (1.f / (float)NN);
}

// ---------------------------------------------------------------------------
// launcher
// ---------------------------------------------------------------------------
extern "C" void kernel_launch(void* const* d_in, const int* in_sizes, int n_in,
                              void* d_out, int out_size, void* d_ws, size_t ws_size,
                              hipStream_t stream) {
  const float* nf    = (const float*)d_in[0];
  const int*   eidx  = (const int*)d_in[1];
  const int*   etype = (const int*)d_in[2];
  const int*   ntype = (const int*)d_in[3];
  const float* ew    = (const float*)d_in[4];
  const float* emb   = (const float*)d_in[5];
  const float* Wself = (const float*)d_in[6];
  const float* bself = (const float*)d_in[7];
  const float* Wrel  = (const float*)d_in[8];
  const float* brel  = (const float*)d_in[9];
  const float* ing   = (const float*)d_in[10];
  const float* inb   = (const float*)d_in[11];
  const float* n1g   = (const float*)d_in[12];
  const float* n1b   = (const float*)d_in[13];
  const float* Wf1   = (const float*)d_in[14];
  const float* bf1   = (const float*)d_in[15];
  const float* Wf2   = (const float*)d_in[16];
  const float* bf2   = (const float*)d_in[17];
  const float* n2g   = (const float*)d_in[18];
  const float* n2b   = (const float*)d_in[19];
  const float* outg  = (const float*)d_in[20];
  const float* outb  = (const float*)d_in[21];
  const float* gml   = (const float*)d_in[22];
  const float* dml   = (const float*)d_in[23];

  char* ws = (char*)d_ws;
  size_t o = 0;
  auto give = [&](size_t bytes) -> char* {
    char* r = ws + o;
    o += (bytes + 255) & ~(size_t)255;
    return r;
  };
  const size_t ND = (size_t)NN * DIM;
  float*     base   = (float*)give(ND * 4);
  float*     x1     = (float*)give(ND * 4);
  float*     selfF  = (float*)give(ND * 4);          // reused as FFN output
  float*     agg    = (float*)give(ND * 4);
  float*     deg    = (float*)give((size_t)NN * 4);
  float*     colsum = (float*)give(DIM * 4);
  char*      zeroEnd = ws + o;                        // agg..colsum contiguous zero span
  _Float16*  xh     = (_Float16*)give(ND * 2);
  _Float16*  x1h    = (_Float16*)give(ND * 2);
  _Float16*  yh     = (_Float16*)give((size_t)NN * 2 * DIM * 2);   // [NN,1024] fp16
  _Float16*  Hh     = (_Float16*)give((size_t)NREL * ND * 2);      // [4,NN,512] fp16
  _Float16*  WrelH  = (_Float16*)give((size_t)NREL * DIM * DIM * 2);
  _Float16*  WselfH = (_Float16*)give((size_t)DIM * DIM * 2);
  _Float16*  Wf1H   = (_Float16*)give((size_t)2 * DIM * DIM * 2);
  _Float16*  Wf2H   = (_Float16*)give((size_t)DIM * 2 * DIM * 2);

  float* outX = (float*)d_out;
  float* outP = outX + ND;

  // 1. zero agg/deg/colsum span
  long zn = (long)((zeroEnd - (char*)agg) / 4);
  zero_f32<<<2048, 256, 0, stream>>>(agg, zn);

  // 2. weight fp32 -> fp16
  cvt_f32_f16<<<512, 256, 0, stream>>>(Wrel,  WrelH,  (long)NREL * DIM * DIM);
  cvt_f32_f16<<<256, 256, 0, stream>>>(Wself, WselfH, (long)DIM * DIM);
  cvt_f32_f16<<<256, 256, 0, stream>>>(Wf1,   Wf1H,   (long)2 * DIM * DIM);
  cvt_f32_f16<<<256, 256, 0, stream>>>(Wf2,   Wf2H,   (long)DIM * 2 * DIM);

  // 3. input LN + type embedding
  prep_ln<<<NN, 256, 0, stream>>>(nf, ntype, emb, ing, inb, base, xh);

  const int GY = (NN + 127) / 128;

  // 4. H_r = X @ W_r^T + b_r  for r=0..3  (batched over grid.z), fp16 out
  gemm_wmma<1><<<dim3(DIM / 64, GY, NREL), 256, 0, stream>>>(
      xh, WrelH, brel, (void*)Hh, NN, DIM, DIM,
      (long)DIM * DIM, (long)DIM, (long)ND);

  // 5. selfF = X @ W_self^T + b_self, fp32 out
  gemm_wmma<0><<<dim3(DIM / 64, GY, 1), 256, 0, stream>>>(
      xh, WselfH, bself, (void*)selfF, NN, DIM, DIM, 0, 0, 0);

  // 6. edge scatter (forward + reverse relations)
  scatter_edges<<<2 * NE, 256, 0, stream>>>(Hh, eidx, etype, ew, agg, deg);

  // 7. combine + LN1
  combine_ln<<<NN, 256, 0, stream>>>(base, selfF, agg, deg, n1g, n1b, x1, x1h);

  // 8. FFN1: gelu(x1 @ Wf1^T + bf1) -> fp16 [NN,1024]
  gemm_wmma<2><<<dim3((2 * DIM) / 64, GY, 1), 256, 0, stream>>>(
      x1h, Wf1H, bf1, (void*)yh, NN, 2 * DIM, DIM, 0, 0, 0);

  // 9. FFN2: y @ Wf2^T + bf2 -> fp32 (reuse selfF)
  gemm_wmma<0><<<dim3(DIM / 64, GY, 1), 256, 0, stream>>>(
      yh, Wf2H, bf2, (void*)selfF, NN, DIM, 2 * DIM, 0, 0, 0);

  // 10. LN2 + out-LN + graph mix -> d_out
  final_ln<<<NN, 256, 0, stream>>>(x1, selfF, base, n2g, n2b, outg, outb, gml, outX);

  // 11. column sums + pooled mix
  colsum_k<<<256, 256, 0, stream>>>(outX, colsum);
  pooled_k<<<1, DIM, 0, stream>>>(outX, colsum, dml, outP);
}
